// MLP_24464133718169
// MI455X (gfx1250) — matrix operand order repaired and checked by reading, not compiled, not run on hardware
//
#include <hip/hip_runtime.h>
#include <hip/hip_bf16.h>

// Problem constants (from reference)
#define BATCH 32
#define LEN   720      // L
#define NV    1024     // V
#define PRED  720      // P
#define NE    8        // experts
#define CAP   (BATCH*NV)   // max tokens per expert list = 32768

#define KPAD  740      // LDS K stride: >=736 (23*32); 740 halves = 370 dwords = 2*odd -> low bank conflicts, 8B-aligned rows
#define KPACK 736      // packed K extent in pre-converted We (bf16), zero-padded 720..735
#define TOKS_PER_BLK 64
#define EXP_THREADS 128    // 4 waves of 32
#define PSTEP 48           // 3 p-subtiles per wave; 720 = 48*15 exactly

typedef __attribute__((ext_vector_type(16))) __bf16 v16bf;
typedef __attribute__((ext_vector_type(8)))  float  v8f;

union FragBF {
    v16bf v;
    uint2 d[4];
};
static_assert(sizeof(v16bf) == 32, "v16bf must be 32 bytes");

__device__ __forceinline__ unsigned short f2bf(float f) {
    // round-to-nearest-even f32 -> bf16 bits
    unsigned int u = __float_as_uint(f);
    u += 0x7FFFu + ((u >> 16) & 1u);
    return (unsigned short)(u >> 16);
}

// Async DMA: copy 8 bytes global -> LDS, tracked by ASYNCcnt (CDNA5 §15.18.3 op 97)
__device__ __forceinline__ void asyncCopyB64(const void* g, void* lds) {
    unsigned loff = (unsigned)(uintptr_t)lds;   // LDS aperture: low 32 bits = wave-relative LDS byte offset
    asm volatile("global_load_async_to_lds_b64 %0, %1, off"
                 :: "v"(loff), "v"(g) : "memory");
}
__device__ __forceinline__ void waitAsync0() {
    asm volatile("s_wait_asynccnt 0x0" ::: "memory");
}

// A 16x32 bf16 fragment: lanes 0-15 hold K{0..7,16..23}, lanes 16-31 K{8..15,24..31}
__device__ __forceinline__ void loadFragA(const unsigned short* rowBase, int k0, int hi, FragBF& f) {
    const unsigned short* p = rowBase + k0 + hi * 8;
    f.d[0] = *(const uint2*)(p);
    f.d[1] = *(const uint2*)(p + 4);
    f.d[2] = *(const uint2*)(p + 16);
    f.d[3] = *(const uint2*)(p + 20);
}

// B 32x16 bf16 fragment: lanes 0-15 hold K{0..7,16..23}, lanes 16-31 K{8..15,24..31} of column ln
__device__ __forceinline__ void loadFragB(const unsigned short* rowBase, int k0, int hi, FragBF& f) {
    const unsigned short* p = rowBase + k0 + hi * 8;
    f.d[0] = *(const uint2*)(p);
    f.d[1] = *(const uint2*)(p + 4);
    f.d[2] = *(const uint2*)(p + 16);
    f.d[3] = *(const uint2*)(p + 20);
}

// ---------------------------------------------------------------------------
// Kernel 0: zero the main output region (scatter target) + routing counters
// ---------------------------------------------------------------------------
__global__ __launch_bounds__(256) void kzero(float4* __restrict__ out4, long n4,
                                             int* __restrict__ cnt) {
    long i = (long)blockIdx.x * 256 + threadIdx.x;
    if (i < n4) out4[i] = make_float4(0.f, 0.f, 0.f, 0.f);
    if (blockIdx.x == 0 && threadIdx.x < NE) cnt[threadIdx.x] = 0;
}

// ---------------------------------------------------------------------------
// Kernel 0b: pre-convert We (f32) -> bf16 rows padded to KPACK (zeros 720..735)
// so expert A-tiles become raw byte-copyable by the async DMA path.
// ---------------------------------------------------------------------------
__global__ __launch_bounds__(256) void kconv(const float* __restrict__ We,
                                             unsigned short* __restrict__ Web) {
    long i = (long)blockIdx.x * 256 + threadIdx.x;
    const long total = (long)NE * PRED * KPACK;
    if (i >= total) return;
    long pe = i / KPACK;                 // e*PRED + p
    int  k  = (int)(i - pe * KPACK);
    float v = (k < LEN) ? We[pe * LEN + k] : 0.f;
    Web[i] = f2bf(v);
}

// ---------------------------------------------------------------------------
// Kernel 1: gating logits -> softmax -> top-2 routing lists (+ gate buffer)
// ---------------------------------------------------------------------------
__global__ __launch_bounds__(256) void kgate(const float* __restrict__ x,
                                             const float* __restrict__ Wg,
                                             float* __restrict__ gateb,
                                             int*  __restrict__ cnt,
                                             int*  __restrict__ idxL,
                                             float* __restrict__ wtL) {
    __shared__ float WgS[NE * LEN];   // 23 KB
    for (int i = threadIdx.x; i < NE * LEN; i += 256) WgS[i] = Wg[i];
    __syncthreads();

    int t = blockIdx.x * 256 + threadIdx.x;   // token id = b*V + v
    int b = t >> 10;            // V = 1024
    int v = t & (NV - 1);

    const float* xp = x + (size_t)b * LEN * NV + v;
    float acc[NE];
    #pragma unroll
    for (int e = 0; e < NE; ++e) acc[e] = 0.f;

    for (int l = 0; l < LEN; ++l) {
        float xv = xp[(size_t)l * NV];          // coalesced across lanes (v)
        #pragma unroll
        for (int e = 0; e < NE; ++e) acc[e] += xv * WgS[e * LEN + l];  // LDS broadcast
    }

    // softmax over 8
    float mx = acc[0];
    #pragma unroll
    for (int e = 1; e < NE; ++e) mx = fmaxf(mx, acc[e]);
    float s = 0.f;
    #pragma unroll
    for (int e = 0; e < NE; ++e) { acc[e] = __expf(acc[e] - mx); s += acc[e]; }
    float inv = 1.f / s;
    #pragma unroll
    for (int e = 0; e < NE; ++e) acc[e] *= inv;

    float* gb = gateb + (size_t)t * NE;
    #pragma unroll
    for (int e = 0; e < NE; ++e) gb[e] = acc[e];

    // top-2 (ties -> lower index, matching lax.top_k)
    int e1 = 0; float w1 = acc[0];
    #pragma unroll
    for (int e = 1; e < NE; ++e) if (acc[e] > w1) { w1 = acc[e]; e1 = e; }
    int e2 = (e1 == 0) ? 1 : 0; float w2 = acc[e2];
    #pragma unroll
    for (int e = 0; e < NE; ++e)
        if (e != e1 && acc[e] > w2) { w2 = acc[e]; e2 = e; }

    int p1 = atomicAdd(&cnt[e1], 1);
    idxL[e1 * CAP + p1] = t;  wtL[e1 * CAP + p1] = w1;
    int p2 = atomicAdd(&cnt[e2], 1);
    idxL[e2 * CAP + p2] = t;  wtL[e2 * CAP + p2] = w2;
}

// ---------------------------------------------------------------------------
// Kernel 2: gate.mean(axis=0) -> out[B*P*V + v*E + e], deterministic
// ---------------------------------------------------------------------------
__global__ __launch_bounds__(256) void kgmean(const float* __restrict__ gateb,
                                              float* __restrict__ outg) {
    int i = blockIdx.x * 256 + threadIdx.x;
    if (i >= NV * NE) return;
    int v = i >> 3, e = i & (NE - 1);
    float s = 0.f;
    for (int b = 0; b < BATCH; ++b)
        s += gateb[((size_t)b * NV + v) * NE + e];
    outg[v * NE + e] = s * (1.f / BATCH);
}

// ---------------------------------------------------------------------------
// Kernel 3: routed expert GEMM with bf16 WMMA + async-DMA A-tile staging.
// grid = (NE, CAP/TOKS_PER_BLK). Block: 4 waves, 64 tokens of expert e's list.
// LDS: Xs[64][KPAD] bf16 (gathered token columns), As0/As1[48][KPAD] bf16
// (double-buffered expert rows, filled by GLOBAL_LOAD_ASYNC_TO_LDS_B64 from
// the pre-converted bf16 We -- next p-step's tile streams in while the WMMA
// loop runs; consumer sync = s_wait_asynccnt 0 + workgroup barrier).
// Per K-step each wave: 3 A-frags + 1 B-frag feed 3 WMMAs; fragments are
// register double-buffered (fully unrolled K loop) -> partial s_wait_dscnt.
// Output scattered with atomicAdd(f32): exactly 2 adds per element (top-2);
// two-addend FP add onto 0 is order-invariant -> deterministic.
// ---------------------------------------------------------------------------
__global__ __launch_bounds__(EXP_THREADS) void kexpert(
        const float* __restrict__ x,  const unsigned short* __restrict__ Web,
        const float* __restrict__ be, const int* __restrict__ cnt,
        const int* __restrict__ idxL, const float* __restrict__ wtL,
        float* __restrict__ out) {
    extern __shared__ char smemRaw[];
    unsigned short* Xs  = (unsigned short*)smemRaw;          // [TOKS_PER_BLK][KPAD]
    unsigned short* As0 = Xs + TOKS_PER_BLK * KPAD;          // [PSTEP][KPAD]
    unsigned short* As1 = As0 + PSTEP * KPAD;                // [PSTEP][KPAD]
    __shared__ int   tokId[TOKS_PER_BLK];
    __shared__ float tokW[TOKS_PER_BLK];

    const int e  = blockIdx.x;
    const int n  = cnt[e];
    const int t0 = blockIdx.y * TOKS_PER_BLK;
    if (t0 >= n) return;                        // block-uniform
    const int m  = min(TOKS_PER_BLK, n - t0);
    const int tid = threadIdx.x;

    if (tid < TOKS_PER_BLK) {
        if (tid < m) {
            tokId[tid] = idxL[e * CAP + t0 + tid];
            tokW[tid]  = wtL[e * CAP + t0 + tid];
        } else {
            tokId[tid] = 0; tokW[tid] = 0.f;    // zero-weight pad
        }
    }
    __syncthreads();

    // Issue async DMA for the first A tile before doing the Xs gather, so the
    // DMA overlaps the (VALU-heavy) gather/convert phase.
    const unsigned short* webE = Web + (size_t)e * PRED * KPACK;
    {
        const unsigned short* src = webE;       // p0 = 0
        for (int i = tid; i < PSTEP * (KPACK / 4); i += EXP_THREADS) {
            int row = i / (KPACK / 4);
            int ch  = i - row * (KPACK / 4);
            asyncCopyB64(src + (size_t)row * KPACK + ch * 4,
                         As0 + row * KPAD + ch * 4);
        }
    }

    // Gather token columns -> Xs (bf16). Lanes span tokens at fixed k.
    for (int i = tid; i < TOKS_PER_BLK * KPAD; i += EXP_THREADS) {
        int j = i & (TOKS_PER_BLK - 1);
        int k = i >> 6;                          // TOKS_PER_BLK == 64
        float vx = 0.f;
        if (j < m && k < LEN) {
            int tok = tokId[j];
            int b = tok >> 10, v = tok & (NV - 1);
            vx = x[((size_t)b * LEN + k) * NV + v];
        }
        Xs[j * KPAD + k] = f2bf(vx);
    }
    __syncthreads();

    const int lane = tid & 31;
    const int wave = tid >> 5;                   // 0..3 -> token tile base wave*16
    const int hi   = lane >> 4;
    const int ln   = lane & 15;

    const unsigned short* bbase = Xs + (wave * 16 + ln) * KPAD;

    const int jtok = wave * 16 + ln;
    const int tok  = tokId[jtok];
    const float w  = tokW[jtok];
    const int bI = tok >> 10, vI = tok & (NV - 1);

    for (int pt = 0; pt < PRED / PSTEP; ++pt) {
        const int p0 = pt * PSTEP;
        unsigned short* curAs = (pt & 1) ? As1 : As0;
        unsigned short* nxtAs = (pt & 1) ? As0 : As1;

        // A tile for this p-step must have landed; all waves' DMA done.
        waitAsync0();
        __syncthreads();

        // Kick off DMA for the next p-step's A tile, then compute on current.
        if (pt + 1 < PRED / PSTEP) {
            const unsigned short* src = webE + (size_t)(p0 + PSTEP) * KPACK;
            for (int i = tid; i < PSTEP * (KPACK / 4); i += EXP_THREADS) {
                int row = i / (KPACK / 4);
                int ch  = i - row * (KPACK / 4);
                asyncCopyB64(src + (size_t)row * KPACK + ch * 4,
                             nxtAs + row * KPAD + ch * 4);
            }
        }

        const unsigned short* a0base = curAs + (0 * 16 + ln) * KPAD;
        const unsigned short* a1base = curAs + (1 * 16 + ln) * KPAD;
        const unsigned short* a2base = curAs + (2 * 16 + ln) * KPAD;

        v8f acc0 = {0.f,0.f,0.f,0.f,0.f,0.f,0.f,0.f};
        v8f acc1 = {0.f,0.f,0.f,0.f,0.f,0.f,0.f,0.f};
        v8f acc2 = {0.f,0.f,0.f,0.f,0.f,0.f,0.f,0.f};

        // software-pipelined, register double-buffered K loop (fully unrolled)
        FragBF A0[2], A1[2], A2[2], Bf[2];
        loadFragA(a0base, 0, hi, A0[0]);
        loadFragA(a1base, 0, hi, A1[0]);
        loadFragA(a2base, 0, hi, A2[0]);
        loadFragB(bbase,  0, hi, Bf[0]);

        #pragma unroll
        for (int ks = 0; ks < 23; ++ks) {
            const int cur = ks & 1, nxt = cur ^ 1;
            if (ks < 22) {
                const int k0 = (ks + 1) * 32;
                loadFragA(a0base, k0, hi, A0[nxt]);
                loadFragA(a1base, k0, hi, A1[nxt]);
                loadFragA(a2base, k0, hi, A2[nxt]);
                loadFragB(bbase,  k0, hi, Bf[nxt]);
            }
            acc0 = __builtin_amdgcn_wmma_f32_16x16x32_bf16(
                false, A0[cur].v, false, Bf[cur].v, (short)0, acc0, false, false);
            acc1 = __builtin_amdgcn_wmma_f32_16x16x32_bf16(
                false, A1[cur].v, false, Bf[cur].v, (short)0, acc1, false, false);
            acc2 = __builtin_amdgcn_wmma_f32_16x16x32_bf16(
                false, A2[cur].v, false, Bf[cur].v, (short)0, acc2, false, false);
        }

        // D layout: element r -> row M = r + 8*hi, col N = ln (token)
        const float* bep0 = be + e * PRED + p0 + 8 * hi;
        float* op0 = out + ((size_t)bI * PRED + p0 + 8 * hi) * NV + vI;
        #pragma unroll
        for (int r = 0; r < 8; ++r)
            atomicAdd(op0 + (size_t)r * NV, w * (acc0[r] + bep0[r]));
        const float* bep1 = bep0 + 16;
        float* op1 = op0 + (size_t)16 * NV;
        #pragma unroll
        for (int r = 0; r < 8; ++r)
            atomicAdd(op1 + (size_t)r * NV, w * (acc1[r] + bep1[r]));
        const float* bep2 = bep0 + 32;
        float* op2 = op0 + (size_t)32 * NV;
        #pragma unroll
        for (int r = 0; r < 8; ++r)
            atomicAdd(op2 + (size_t)r * NV, w * (acc2[r] + bep2[r]));
        // No trailing barrier needed: next iteration's waitAsync0+barrier
        // orders DMA-into-curAs (issued at pt+1) after all reads at pt.
    }
}

// ---------------------------------------------------------------------------
extern "C" void kernel_launch(void* const* d_in, const int* in_sizes, int n_in,
                              void* d_out, int out_size, void* d_ws, size_t ws_size,
                              hipStream_t stream) {
    const float* x  = (const float*)d_in[0];   // [B, L, V]
    const float* Wg = (const float*)d_in[1];   // [E, L]
    const float* We = (const float*)d_in[2];   // [E, P, L]
    const float* be = (const float*)d_in[3];   // [E, P]
    float* out = (float*)d_out;                // [B,P,V] ++ [V,E]

    // workspace layout
    int*   cnt   = (int*)d_ws;                                          // 8 ints
    float* gateb = (float*)((char*)d_ws + 256);                         // B*V*E f32 (1 MB)
    int*   idxL  = (int*)((char*)gateb + (size_t)BATCH * NV * NE * 4);  // E*CAP int (1 MB)
    float* wtL   = (float*)((char*)idxL + (size_t)NE * CAP * 4);        // E*CAP f32 (1 MB)
    unsigned short* Web = (unsigned short*)((char*)wtL + (size_t)NE * CAP * 4); // E*P*KPACK bf16 (8.5 MB)

    const long mainN  = (long)BATCH * PRED * NV;      // 23,592,960
    const long n4     = mainN / 4;
    float* outg = out + mainN;                        // gate.mean region [V,E]

    // 1) zero scatter target + counters
    kzero<<<dim3((unsigned)((n4 + 255) / 256)), dim3(256), 0, stream>>>(
        (float4*)out, n4, cnt);

    // 1b) pre-convert We -> bf16 (rows padded to KPACK)
    {
        const long totW = (long)NE * PRED * KPACK;
        kconv<<<dim3((unsigned)((totW + 255) / 256)), dim3(256), 0, stream>>>(We, Web);
    }

    // 2) gating + routing
    kgate<<<dim3(BATCH * NV / 256), dim3(256), 0, stream>>>(
        x, Wg, gateb, cnt, idxL, wtL);

    // 3) gate.mean(0) (deterministic)
    kgmean<<<dim3((NV * NE + 255) / 256), dim3(256), 0, stream>>>(gateb, outg);

    // 4) routed expert GEMM (bf16 WMMA + async DMA), 236.8 KB dynamic LDS
    const size_t smem = (size_t)(TOKS_PER_BLK + 2 * PSTEP) * KPAD * sizeof(unsigned short);
    kexpert<<<dim3(NE, CAP / TOKS_PER_BLK), dim3(EXP_THREADS), smem, stream>>>(
        x, Web, be, cnt, idxL, wtL, out);
}
